// VectorQuantizer_27917287424813
// MI455X (gfx1250) — matrix operand order repaired and checked by reading, not compiled, not run on hardware
//
#include <hip/hip_runtime.h>
#include <hip/hip_bf16.h>

#define N_ROWS 32768
#define K_CB   8192
#define DIM    512
#define ND     (N_ROWS * DIM)

#define LSTRIDE 520                 // u16 per LDS row (512 data + 8 pad = 16B)
#define LBUF    (32 * LSTRIDE)      // u16 per buffer (33,280 B)
#define NIT     (K_CB / 32)         // 256 dual-tile iterations

typedef __attribute__((ext_vector_type(16))) __bf16 v16bf;
typedef __attribute__((ext_vector_type(8)))  __bf16 v8bf;
typedef __attribute__((ext_vector_type(8)))  float  v8f;

static __device__ __forceinline__ unsigned short f32_to_bf16_rne(float f) {
    unsigned int u = __float_as_uint(f);
    unsigned int r = u + 0x7FFFu + ((u >> 16) & 1u);
    return (unsigned short)(r >> 16);
}

// One wave per row: L2-normalize a 512-wide f32 row, emit bf16.
__global__ __launch_bounds__(256) void vq_normalize_bf16(const float* __restrict__ src,
                                                         unsigned short* __restrict__ dst,
                                                         int rows) {
    const int wave = threadIdx.x >> 5;
    const int lane = threadIdx.x & 31;
    const int row  = blockIdx.x * 8 + wave;
    if (row >= rows) return;
    const float* s = src + (size_t)row * DIM;
    float4 v[4];
    float acc = 0.f;
#pragma unroll
    for (int t = 0; t < 4; ++t) {
        v[t] = *(const float4*)(s + t * 128 + lane * 4);
        acc += v[t].x * v[t].x + v[t].y * v[t].y + v[t].z * v[t].z + v[t].w * v[t].w;
    }
#pragma unroll
    for (int m = 16; m >= 1; m >>= 1) acc += __shfl_xor(acc, m, 32);
    const float scale = 1.0f / fmaxf(sqrtf(acc), 1e-12f);
    unsigned short* d = dst + (size_t)row * DIM;
#pragma unroll
    for (int t = 0; t < 4; ++t) {
        ushort4 o;
        o.x = f32_to_bf16_rne(v[t].x * scale);
        o.y = f32_to_bf16_rne(v[t].y * scale);
        o.z = f32_to_bf16_rne(v[t].z * scale);
        o.w = f32_to_bf16_rne(v[t].w * scale);
        *(ushort4*)(d + t * 128 + lane * 4) = o;
    }
}

// LDS-staged GEMM-argmax. Block = 8 waves x 32 rows = 256 rows. Per iteration
// the block stages one 32-col x 512-K bf16 codebook tile-pair (32 KB) into a
// double-buffered, row-padded LDS buffer; all 8 waves consume it via
// ds_load_b128 (base VGPR + immediate offsets only) -> v_wmma. Each chunk
// feeds 2 WMMAs (two register-resident A tiles); dual N chains -> 4 chains.
// The kk chain uses an explicitly alternating two-buffer scheme so chunk kk+1
// loads while chunk kk's 4 WMMAs execute (LDS latency fully hidden).
__global__ __launch_bounds__(256) void vq_argmax_wmma(const unsigned short* __restrict__ xnb,
                                                      const unsigned short* __restrict__ wnb,
                                                      int* __restrict__ idx_out,
                                                      float* __restrict__ idx_out_f) {
    __shared__ unsigned short lds[2 * LBUF];          // 66,560 B

    const int tid  = threadIdx.x;
    const int wave = tid >> 5;
    const int lane = tid & 31;
    const int m    = lane & 15;           // A row / B col within tile
    const int hi   = lane >> 4;           // half-wave selects K sub-range
    const int row0 = blockIdx.x * 256 + wave * 32;

    // A tiles in ISA 16-bit A layout (16x32 per chunk):
    // lanes 0-15: K {0..7,16..23}; lanes 16-31: K {8..15,24..31} (+32*kk).
    v16bf a0[16], a1[16];
    {
        const unsigned short* xr0 = xnb + (size_t)(row0 + m) * DIM;
        const unsigned short* xr1 = xr0 + 16 * DIM;
#pragma unroll
        for (int kk = 0; kk < 16; ++kk) {
            const int base = kk * 32 + hi * 8;
            v8bf lo0 = *(const v8bf*)(xr0 + base);
            v8bf hi0 = *(const v8bf*)(xr0 + base + 16);
            v8bf lo1 = *(const v8bf*)(xr1 + base);
            v8bf hi1 = *(const v8bf*)(xr1 + base + 16);
            v16bf av0, av1;
#pragma unroll
            for (int i = 0; i < 8; ++i) {
                av0[i] = lo0[i]; av0[i + 8] = hi0[i];
                av1[i] = lo1[i]; av1[i + 8] = hi1[i];
            }
            a0[kk] = av0;
            a1[kk] = av1;
        }
    }

    float run_v0[8], run_v1[8];
    int   run_c0[8], run_c1[8];
#pragma unroll
    for (int j = 0; j < 8; ++j) {
        run_v0[j] = -2.0f; run_c0[j] = 0x7fffffff;
        run_v1[j] = -2.0f; run_c1[j] = 0x7fffffff;
    }

    // Staging: granule G = tid + s*256 -> c = (tid>>6) + 4s, g = tid&63.
    // Global u16 offset = it*16384 + c*512 + g*8  = gbase + it*16384 + s*2048.
    // LDS    u16 offset =             c*520 + g*8 = lbase + s*2080.
    const int gbase = ((tid >> 6) << 9) + ((tid & 63) << 3);
    const int lbase = (tid >> 6) * LSTRIDE + ((tid & 63) << 3);
    v8bf sreg[8];

    auto stage_load = [&](int it) {
        const unsigned short* gp = wnb + ((size_t)it << 14) + gbase;
#pragma unroll
        for (int s = 0; s < 8; ++s) sreg[s] = *(const v8bf*)(gp + s * 2048);
    };
    auto stage_store = [&](int buf) {
        unsigned short* sp = &lds[buf * LBUF + lbase];
#pragma unroll
        for (int s = 0; s < 8; ++s) *(v8bf*)(sp + s * 2080) = sreg[s];
    };

    // Per-lane compute base pointers (include hi half-K offset); all ds_load
    // addresses are base + compile-time immediate (kk*64 [+32] bytes).
    const unsigned short* pb0 = &lds[m * LSTRIDE + hi * 8];
    const unsigned short* pb1 = pb0 + 16 * LSTRIDE;

    stage_load(0);
    stage_store(0);
    __syncthreads();

    for (int it = 0; it < NIT; ++it) {
        const int cur = it & 1;
        const bool more = (it + 1 < NIT);
        if (more) stage_load(it + 1);         // global loads fly during compute

        const unsigned short* p0c = pb0 + cur * LBUF;
        const unsigned short* p1c = pb1 + cur * LBUF;
        v8f c00 = {}, c01 = {}, c10 = {}, c11 = {};

        // Explicit two-buffer LDS pipeline: buffers alternate by kk parity, so
        // both are simultaneously live and cannot be coalesced into one.
        v8bf blo0[2], bhi0[2], blo1[2], bhi1[2];
        blo0[0] = *(const v8bf*)(p0c);
        bhi0[0] = *(const v8bf*)(p0c + 16);
        blo1[0] = *(const v8bf*)(p1c);
        bhi1[0] = *(const v8bf*)(p1c + 16);
#pragma unroll
        for (int kk = 0; kk < 16; ++kk) {
            const int cb = kk & 1;
            const int nb = cb ^ 1;
            if (kk < 15) {                    // prefetch chunk kk+1 into nb
                blo0[nb] = *(const v8bf*)(p0c + (kk + 1) * 32);
                bhi0[nb] = *(const v8bf*)(p0c + (kk + 1) * 32 + 16);
                blo1[nb] = *(const v8bf*)(p1c + (kk + 1) * 32);
                bhi1[nb] = *(const v8bf*)(p1c + (kk + 1) * 32 + 16);
            }
            v16bf b0, b1;
#pragma unroll
            for (int i = 0; i < 8; ++i) {
                b0[i] = blo0[cb][i]; b0[i + 8] = bhi0[cb][i];
                b1[i] = blo1[cb][i]; b1[i + 8] = bhi1[cb][i];
            }
            c00 = __builtin_amdgcn_wmma_f32_16x16x32_bf16(false, a0[kk], false, b0,
                                                          (short)0, c00, false, false);
            c10 = __builtin_amdgcn_wmma_f32_16x16x32_bf16(false, a1[kk], false, b0,
                                                          (short)0, c10, false, false);
            c01 = __builtin_amdgcn_wmma_f32_16x16x32_bf16(false, a0[kk], false, b1,
                                                          (short)0, c01, false, false);
            c11 = __builtin_amdgcn_wmma_f32_16x16x32_bf16(false, a1[kk], false, b1,
                                                          (short)0, c11, false, false);
        }
        // C layout: lane n holds col, VGPR j holds row j + 8*hi.
        const int col0 = it * 32 + m;
        const int col1 = col0 + 16;
#pragma unroll
        for (int j = 0; j < 8; ++j) {
            if (c00[j] > run_v0[j]) { run_v0[j] = c00[j]; run_c0[j] = col0; }
            if (c01[j] > run_v0[j]) { run_v0[j] = c01[j]; run_c0[j] = col1; }
            if (c10[j] > run_v1[j]) { run_v1[j] = c10[j]; run_c1[j] = col0; }
            if (c11[j] > run_v1[j]) { run_v1[j] = c11[j]; run_c1[j] = col1; }
        }

        if (more) stage_store((it + 1) & 1);  // loads have landed by now
        __syncthreads();                      // next buffer ready; cur reads done
    }

    // Final per-row argmax: butterfly across the 16 lanes of each half-wave.
#pragma unroll
    for (int j = 0; j < 8; ++j) {
        float v0 = run_v0[j], v1 = run_v1[j];
        int   i0 = run_c0[j], i1 = run_c1[j];
#pragma unroll
        for (int mm = 8; mm >= 1; mm >>= 1) {
            float o0 = __shfl_xor(v0, mm, 32);
            int   p0 = __shfl_xor(i0, mm, 32);
            if (o0 > v0 || (o0 == v0 && p0 < i0)) { v0 = o0; i0 = p0; }
            float o1 = __shfl_xor(v1, mm, 32);
            int   p1 = __shfl_xor(i1, mm, 32);
            if (o1 > v1 || (o1 == v1 && p1 < i1)) { v1 = o1; i1 = p1; }
        }
        if (m == 0) {                      // lanes 0 and 16 write rows j, j+8
            const int r0 = row0 + hi * 8 + j;
            idx_out[r0]        = i0;
            idx_out_f[r0]      = (float)i0;
            idx_out[r0 + 16]   = i1;
            idx_out_f[r0 + 16] = (float)i1;
        }
    }
}

// One wave per row: quantized_out = normalize(emb[idx]); rowloss = sum (q - xn)^2.
__global__ __launch_bounds__(256) void vq_gather_loss(const float* __restrict__ x,
                                                      const float* __restrict__ w,
                                                      const int* __restrict__ idx,
                                                      float* __restrict__ out_q,
                                                      float* __restrict__ rowloss) {
    const int wave = threadIdx.x >> 5;
    const int lane = threadIdx.x & 31;
    const int row  = blockIdx.x * 8 + wave;
    if (row >= N_ROWS) return;
    const float* xr = x + (size_t)row * DIM;
    const float* wr = w + (size_t)idx[row] * DIM;
    float4 xv[4], wv[4];
    float xacc = 0.f, wacc = 0.f;
#pragma unroll
    for (int t = 0; t < 4; ++t) {
        xv[t] = *(const float4*)(xr + t * 128 + lane * 4);
        wv[t] = *(const float4*)(wr + t * 128 + lane * 4);
        xacc += xv[t].x * xv[t].x + xv[t].y * xv[t].y + xv[t].z * xv[t].z + xv[t].w * xv[t].w;
        wacc += wv[t].x * wv[t].x + wv[t].y * wv[t].y + wv[t].z * wv[t].z + wv[t].w * wv[t].w;
    }
#pragma unroll
    for (int mm = 16; mm >= 1; mm >>= 1) {
        xacc += __shfl_xor(xacc, mm, 32);
        wacc += __shfl_xor(wacc, mm, 32);
    }
    const float xs  = 1.0f / fmaxf(sqrtf(xacc), 1e-12f);
    const float wsc = 1.0f / fmaxf(sqrtf(wacc), 1e-12f);
    float dl = 0.f;
    float* oq = out_q + (size_t)row * DIM;
#pragma unroll
    for (int t = 0; t < 4; ++t) {
        float4 q;
        q.x = wv[t].x * wsc; q.y = wv[t].y * wsc; q.z = wv[t].z * wsc; q.w = wv[t].w * wsc;
        float dx = q.x - xv[t].x * xs, dy = q.y - xv[t].y * xs,
              dz = q.z - xv[t].z * xs, dw = q.w - xv[t].w * xs;
        dl += dx * dx + dy * dy + dz * dz + dw * dw;
        *(float4*)(oq + t * 128 + lane * 4) = q;
    }
#pragma unroll
    for (int mm = 16; mm >= 1; mm >>= 1) dl += __shfl_xor(dl, mm, 32);
    if (lane == 0) rowloss[row] = dl;
}

__global__ __launch_bounds__(256) void vq_finalize(const float* __restrict__ rowloss,
                                                   float* __restrict__ loss_out) {
    __shared__ float sm[256];
    float acc = 0.f;
    for (int i = threadIdx.x; i < N_ROWS; i += 256) acc += rowloss[i];
    sm[threadIdx.x] = acc;
    __syncthreads();
    for (int s = 128; s > 0; s >>= 1) {
        if (threadIdx.x < s) sm[threadIdx.x] += sm[threadIdx.x + s];
        __syncthreads();
    }
    // q_latent_loss == e_latent_loss numerically -> loss = 1.25 * MSE
    if (threadIdx.x == 0) loss_out[0] = 1.25f * sm[0] / (float)ND;
}

extern "C" void kernel_launch(void* const* d_in, const int* in_sizes, int n_in,
                              void* d_out, int out_size, void* d_ws, size_t ws_size,
                              hipStream_t stream) {
    const float* x = (const float*)d_in[0];       // [32768, 512]
    const float* w = (const float*)d_in[1];       // [8192, 512]
    float* out = (float*)d_out;

    char* ws = (char*)d_ws;
    unsigned short* wn_bf16 = (unsigned short*)ws;                                  // 8 MB
    unsigned short* xn_bf16 = (unsigned short*)(ws + (size_t)K_CB * DIM * 2);       // 32 MB
    int*   idx     = (int*)(ws + (size_t)K_CB * DIM * 2 + (size_t)N_ROWS * DIM * 2);
    float* rowloss = (float*)(idx + N_ROWS);

    float* out_q    = out;            // [N, D] quantized_out (== wn[idx])
    float* out_loss = out + ND;       // scalar
    float* out_idx  = out + ND + 1;   // [N] indices (as float)

    vq_normalize_bf16<<<K_CB / 8,    256, 0, stream>>>(w, wn_bf16, K_CB);
    vq_normalize_bf16<<<N_ROWS / 8,  256, 0, stream>>>(x, xn_bf16, N_ROWS);
    vq_argmax_wmma<<<N_ROWS / 256,   256, 0, stream>>>(xn_bf16, wn_bf16, idx, out_idx);
    vq_gather_loss<<<N_ROWS / 8,     256, 0, stream>>>(x, w, idx, out_q, rowloss);
    vq_finalize<<<1,                 256, 0, stream>>>(rowloss, out_loss);
}
